// AGLI_5703716569604
// MI455X (gfx1250) — matrix operand order repaired and checked by reading, not compiled, not run on hardware
//
#include <hip/hip_runtime.h>
#include <hip/hip_bf16.h>

typedef _Float16 v16h __attribute__((ext_vector_type(16)));
typedef _Float16 h8   __attribute__((ext_vector_type(8)));
typedef float    v8f  __attribute__((ext_vector_type(8)));

#define NDIM 4096
static __device__ __forceinline__ float bn_scale(float g) {
    return g * 0.9999950000374997f;  // gamma / sqrt(1 + 1e-5)
}

// ---------------------------------------------------------------------------
// Fragment builders for CDNA5 WMMA 16x16x32 f16 (contiguous k order, same in A and B).
// A (16x32, per lane row=lane&15, kbase=(lane>>4)*16):
//   elements 0..15 <-> K = kbase + i      (contiguous)
// p must point at A_row + kbase (contiguous K-major storage).
__device__ __forceinline__ v16h make_a_frag(const _Float16* p) {
    h8 lo = *(const h8*)p;
    h8 hi = *(const h8*)(p + 8);
    v16h a;
#pragma unroll
    for (int i = 0; i < 8; ++i) { a[i] = lo[i]; a[8 + i] = hi[i]; }
    return a;
}
// B (32x16, per lane col=lane&15, kb=(lane>>4)*16): element i <-> K = kb + i.
// p must point at 16 contiguous K values for this column.
__device__ __forceinline__ v16h make_b_frag(const _Float16* p) {
    h8 lo = *(const h8*)p;
    h8 hi = *(const h8*)(p + 8);
    v16h b;
#pragma unroll
    for (int i = 0; i < 8; ++i) { b[i] = lo[i]; b[8 + i] = hi[i]; }
    return b;
}

// ---------------------------------------------------------------------------
// Generic 3x3 conv (pad 1) + optional bias + BN(eval) + ReLU + optional residual.
// First Gpre input channels are spatially-constant values from gpre[B,Gpre]
// (global-context broadcast concat), remaining Cin channels come from x.
__global__ void bcr_conv(const float* __restrict__ x, const float* __restrict__ w,
                         const float* __restrict__ bias, const float* __restrict__ g,
                         const float* __restrict__ gpre, const float* __restrict__ residual,
                         float* __restrict__ out,
                         int Bn, int Cin, int Gpre, int Cout, int Hh, int Ww) {
    int tid = blockIdx.x * 256 + threadIdx.x;
    int Npix = Hh * Ww;
    int total = Bn * Cout * Npix;
    if (tid >= total) return;
    int p  = tid % Npix;
    int t2 = tid / Npix;
    int co = t2 % Cout;
    int b  = t2 / Cout;
    int py = p / Ww, px = p % Ww;
    int Ct = Gpre + Cin;
    float sum = bias ? bias[co] : 0.0f;
    for (int ci = 0; ci < Ct; ++ci) {
        const float* wr = w + (long long)(co * Ct + ci) * 9;
        if (ci < Gpre) {
            float gv = gpre[b * Gpre + ci];
#pragma unroll
            for (int ky = 0; ky < 3; ++ky) {
                int yy = py + ky - 1;
                if ((unsigned)yy >= (unsigned)Hh) continue;
#pragma unroll
                for (int kx = 0; kx < 3; ++kx) {
                    int xx = px + kx - 1;
                    if ((unsigned)xx >= (unsigned)Ww) continue;
                    sum = fmaf(gv, wr[ky * 3 + kx], sum);
                }
            }
        } else {
            const float* xb = x + (long long)(b * Cin + (ci - Gpre)) * Npix;
#pragma unroll
            for (int ky = 0; ky < 3; ++ky) {
                int yy = py + ky - 1;
                if ((unsigned)yy >= (unsigned)Hh) continue;
#pragma unroll
                for (int kx = 0; kx < 3; ++kx) {
                    int xx = px + kx - 1;
                    if ((unsigned)xx >= (unsigned)Ww) continue;
                    sum = fmaf(xb[yy * Ww + xx], wr[ky * 3 + kx], sum);
                }
            }
        }
    }
    float val = fmaf(sum, bn_scale(g[co]), g[Cout + co]);
    val = fmaxf(val, 0.0f);
    if (residual) val += residual[tid];
    out[tid] = val;
}

// ---------------------------------------------------------------------------
// Global average pool: one block per (b,c), 4096 pixels -> mean. Writes to ws
// vector and directly into the d_out tail (g_s / g_c outputs).
__global__ void gap_kernel(const float* __restrict__ y, float* __restrict__ gvec,
                           float* __restrict__ gout) {
    __shared__ float red[256];
    int bc = blockIdx.x;
    const float* p = y + (long long)bc * NDIM;
    float s = 0.0f;
    for (int i = threadIdx.x; i < NDIM; i += 256) s += p[i];
    red[threadIdx.x] = s;
    __syncthreads();
    for (int o = 128; o > 0; o >>= 1) {
        if (threadIdx.x < o) red[threadIdx.x] += red[threadIdx.x + o];
        __syncthreads();
    }
    if (threadIdx.x == 0) {
        float m = red[0] * (1.0f / (float)NDIM);
        gvec[bc] = m;
        gout[bc] = m;
    }
}

// f32 -> f16 straight cast (layout preserved)
__global__ void cvt_f16(const float* __restrict__ in, _Float16* __restrict__ out, int n) {
    int i = blockIdx.x * 256 + threadIdx.x;
    if (i < n) out[i] = (_Float16)in[i];
}

// fs [B,32,N] f32 -> fsT [B,N,32] f16 (K-major rows for WMMA fragments)
__global__ void transpose_f16(const float* __restrict__ in, _Float16* __restrict__ out) {
    int i = blockIdx.x * 256 + threadIdx.x;  // B*N*32 = 262144 threads
    int c = i & 31;
    int n = (i >> 5) & (NDIM - 1);
    int b = i >> 17;
    out[i] = (_Float16)in[((long long)(b * 32 + c)) * NDIM + n];
}

// ---------------------------------------------------------------------------
// att[b,n,m] = sum_c fs[b,c,n] * fc[b,c,m]   (K = 32, one WMMA per 16x16 tile)
// One wave computes a 16(n) x 128(m) strip: load A once, 8 B tiles.
__global__ void att_gemm(const _Float16* __restrict__ fsT, const _Float16* __restrict__ fcT,
                         float* __restrict__ att) {
    int wid  = (blockIdx.x << 3) + (threadIdx.x >> 5);
    int lane = threadIdx.x & 31;
    int b  = wid >> 13;          // 8192 waves per batch
    int r  = wid & 8191;
    int nt = r >> 5;             // n tile 0..255
    int ms = r & 31;             // m strip 0..31
    int n0 = nt << 4, m0 = ms << 7;
    int row = lane & 15, hi = lane >> 4;
    const long long fb = (long long)b << 17;   // N*32
    v16h a = make_a_frag(fsT + fb + (n0 + row) * 32 + hi * 16);
    int col = row, kb = hi << 4;
    long long ob = (long long)b << 24;         // N*N
#pragma unroll
    for (int t = 0; t < 8; ++t) {
        int m = m0 + (t << 4);
        v16h bf = make_b_frag(fcT + fb + (m + col) * 32 + kb);
        v8f acc;
#pragma unroll
        for (int i = 0; i < 8; ++i) acc[i] = 0.0f;
        acc = __builtin_amdgcn_wmma_f32_16x16x32_f16(false, a, false, bf,
                                                     (short)0, acc, false, false);
#pragma unroll
        for (int rr = 0; rr < 8; ++rr) {
            int nn = n0 + rr + (hi << 3);
            att[ob + (long long)nn * NDIM + m + col] = acc[rr];
        }
    }
}

// ---------------------------------------------------------------------------
// Column stats (softmax over axis -2): one thread per column m.
__global__ void col_stats(const float* __restrict__ att, float* __restrict__ cmax,
                          float* __restrict__ csum) {
    int i = blockIdx.x * 256 + threadIdx.x;   // B*N = 8192
    int b = i >> 12, m = i & (NDIM - 1);
    const float* base = att + ((long long)b << 24) + m;
    float mx = -3.4e38f;
    for (int n = 0; n < NDIM; ++n) mx = fmaxf(mx, base[(long long)n * NDIM]);
    float s = 0.0f;
    for (int n = 0; n < NDIM; ++n) s += __expf(base[(long long)n * NDIM] - mx);
    cmax[i] = mx;
    csum[i] = s;
}

// Row stats (softmax over axis -1): one block per row.
__global__ void row_stats(const float* __restrict__ att, float* __restrict__ rmax,
                          float* __restrict__ rsum) {
    __shared__ float red[256];
    int b = blockIdx.x >> 12, j = blockIdx.x & (NDIM - 1);
    const float* p = att + ((long long)b << 24) + (long long)j * NDIM;
    float mx = -3.4e38f;
    for (int m = threadIdx.x; m < NDIM; m += 256) mx = fmaxf(mx, p[m]);
    red[threadIdx.x] = mx;
    __syncthreads();
    for (int o = 128; o > 0; o >>= 1) {
        if (threadIdx.x < o) red[threadIdx.x] = fmaxf(red[threadIdx.x], red[threadIdx.x + o]);
        __syncthreads();
    }
    float mxa = red[0];
    __syncthreads();
    float s = 0.0f;
    for (int m = threadIdx.x; m < NDIM; m += 256) s += __expf(p[m] - mxa);
    red[threadIdx.x] = s;
    __syncthreads();
    for (int o = 128; o > 0; o >>= 1) {
        if (threadIdx.x < o) red[threadIdx.x] += red[threadIdx.x + o];
        __syncthreads();
    }
    if (threadIdx.x == 0) { rmax[blockIdx.x] = mxa; rsum[blockIdx.x] = red[0]; }
}

// Stats of symmetric att2 = att + att^T (col softmax == row softmax^T).
__global__ void sym_stats(const float* __restrict__ att, float* __restrict__ smax,
                          float* __restrict__ ssum) {
    __shared__ float red[256];
    int b = blockIdx.x >> 12, j = blockIdx.x & (NDIM - 1);
    const long long ab = (long long)b << 24;
    const float* prow = att + ab + (long long)j * NDIM;
    float mx = -3.4e38f;
    for (int n = threadIdx.x; n < NDIM; n += 256)
        mx = fmaxf(mx, prow[n] + att[ab + (long long)n * NDIM + j]);
    red[threadIdx.x] = mx;
    __syncthreads();
    for (int o = 128; o > 0; o >>= 1) {
        if (threadIdx.x < o) red[threadIdx.x] = fmaxf(red[threadIdx.x], red[threadIdx.x + o]);
        __syncthreads();
    }
    float mxa = red[0];
    __syncthreads();
    float s = 0.0f;
    for (int n = threadIdx.x; n < NDIM; n += 256)
        s += __expf(prow[n] + att[ab + (long long)n * NDIM + j] - mxa);
    red[threadIdx.x] = s;
    __syncthreads();
    for (int o = 128; o > 0; o >>= 1) {
        if (threadIdx.x < o) red[threadIdx.x] += red[threadIdx.x + o];
        __syncthreads();
    }
    if (threadIdx.x == 0) { smax[blockIdx.x] = mxa; ssum[blockIdx.x] = red[0]; }
}

// ---------------------------------------------------------------------------
// Fused softmax-GEMM: out[c,col] = (sum_k A[c,k] * exp(att[..]-max[col])) / sum[col]
// TRANSB=false: att[k, col] with col-stats  (outs = xs1 @ sm_cols)
// TRANSB=true : att[col, k] with row-stats  (outc = xc1 @ sm_rows^T-style)
// One wave per 16 output columns; CT 16-row channel tiles; K loop over N in 32s.
template <int CT, bool TRANSB>
__global__ void out_gemm(const _Float16* __restrict__ Ah, const float* __restrict__ att,
                         const float* __restrict__ smax, const float* __restrict__ ssum,
                         float* __restrict__ outb) {
    int wid  = (blockIdx.x << 3) + (threadIdx.x >> 5);   // B*256 waves
    int lane = threadIdx.x & 31;
    int b  = wid >> 8;
    int mt = wid & 255;
    int m0 = mt << 4;
    int row = lane & 15, hi = lane >> 4;
    int col = row, kb = hi << 4, kbA = hi << 4;
    const long long ab = (long long)b << 24;
    float bmax = smax[(b << 12) + m0 + col];
    v8f acc[CT];
#pragma unroll
    for (int ct = 0; ct < CT; ++ct)
#pragma unroll
        for (int i = 0; i < 8; ++i) acc[ct][i] = 0.0f;

    for (int kc = 0; kc < NDIM / 32; ++kc) {
        int k0 = kc << 5;
        v16h bf;
#pragma unroll
        for (int i = 0; i < 16; ++i) {
            int k = k0 + kb + i;
            float v = TRANSB ? att[ab + (long long)(m0 + col) * NDIM + k]
                             : att[ab + (long long)k * NDIM + m0 + col];
            bf[i] = (_Float16)__expf(v - bmax);
        }
#pragma unroll
        for (int ct = 0; ct < CT; ++ct) {
            const _Float16* ar =
                Ah + (long long)(b * CT * 16 + ct * 16 + row) * NDIM + k0 + kbA;
            v16h a = make_a_frag(ar);
            acc[ct] = __builtin_amdgcn_wmma_f32_16x16x32_f16(false, a, false, bf,
                                                             (short)0, acc[ct], false, false);
        }
    }
    float inv = 1.0f / ssum[(b << 12) + m0 + col];
#pragma unroll
    for (int ct = 0; ct < CT; ++ct)
#pragma unroll
        for (int rr = 0; rr < 8; ++rr) {
            int c = ct * 16 + rr + (hi << 3);
            outb[(long long)(b * CT * 16 + c) * NDIM + m0 + col] = acc[ct][rr] * inv;
        }
}

// ---------------------------------------------------------------------------
// In-place att -> att3 using symmetry: process tile pair (i,j)/(j,i) together.
__global__ void att3_inplace(float* __restrict__ att, const float* __restrict__ smax,
                             const float* __restrict__ ssum) {
    int bx = blockIdx.x, by = blockIdx.y, b = blockIdx.z;
    if (bx < by) return;   // uniform per block
    __shared__ float sA[16][17], sB[16][17];
    long long base = (long long)b << 24;
    int i0 = by << 4, j0 = bx << 4;
    int tx = threadIdx.x, ty = threadIdx.y;
    sA[ty][tx] = att[base + (long long)(i0 + ty) * NDIM + j0 + tx];
    sB[ty][tx] = att[base + (long long)(j0 + ty) * NDIM + i0 + tx];
    __syncthreads();
    {
        int n = i0 + ty, m = j0 + tx;
        float a2 = sA[ty][tx] + sB[tx][ty];
        float v = __expf(a2 - smax[(b << 12) + m]) / ssum[(b << 12) + m] +
                  __expf(a2 - smax[(b << 12) + n]) / ssum[(b << 12) + n];
        att[base + (long long)n * NDIM + m] = v;
    }
    if (bx != by) {
        int n = j0 + ty, m = i0 + tx;
        float a2 = sB[ty][tx] + sA[tx][ty];
        float v = __expf(a2 - smax[(b << 12) + m]) / ssum[(b << 12) + m] +
                  __expf(a2 - smax[(b << 12) + n]) / ssum[(b << 12) + n];
        att[base + (long long)n * NDIM + m] = v;
    }
}

// ---------------------------------------------------------------------------
extern "C" void kernel_launch(void* const* d_in, const int* in_sizes, int n_in,
                              void* d_out, int out_size, void* d_ws, size_t ws_size,
                              hipStream_t stream) {
    (void)in_sizes; (void)n_in; (void)out_size; (void)ws_size;
    const float* xs   = (const float*)d_in[0];
    const float* xc   = (const float*)d_in[1];
    const float* w_g1 = (const float*)d_in[2];
    const float* g_g1 = (const float*)d_in[3];
    const float* w_g2 = (const float*)d_in[4];
    const float* g_g2 = (const float*)d_in[5];
    const float* w_ls = (const float*)d_in[6];
    const float* b_ls = (const float*)d_in[7];
    const float* g_ls = (const float*)d_in[8];
    const float* w_lc = (const float*)d_in[9];
    const float* b_lc = (const float*)d_in[10];
    const float* g_lc = (const float*)d_in[11];
    const float* w_c1 = (const float*)d_in[12];
    const float* g_c1 = (const float*)d_in[13];
    const float* w_c2 = (const float*)d_in[14];
    const float* g_c2 = (const float*)d_in[15];
    const float* w_o1 = (const float*)d_in[16];
    const float* g_o1 = (const float*)d_in[17];
    const float* w_o2 = (const float*)d_in[18];
    const float* g_o2 = (const float*)d_in[19];
    const float* w_oa = (const float*)d_in[20];
    const float* g_oa = (const float*)d_in[21];
    float* out = (float*)d_out;

    // d_out layout: att_map | out_s | out_c | g_s | g_c
    const long long OFF_OUTS = 33554432LL;           // 2*1*4096*4096
    const long long OFF_OUTC = OFF_OUTS + 524288LL;  // + 2*64*4096
    const long long OFF_GS   = OFF_OUTC + 1048576LL; // + 2*128*4096
    const long long OFF_GC   = OFF_GS + 64LL;

    // Workspace carve-up (f32 region then f16 region, all 16B aligned)
    float* att  = (float*)d_ws;             // 33,554,432
    float* ys   = att + 33554432;           // 262,144  (bcr(xs,w_g1) / later fs)
    float* yc   = ys + 262144;              // 262,144  (bcr(xc,w_g2) / later fc)
    float* xs1  = yc + 262144;              // 524,288
    float* xc1  = xs1 + 524288;             // 1,048,576
    float* outs = xc1 + 1048576;            // 524,288
    float* outc = outs + 524288;            // 1,048,576
    float* gsv  = outc + 1048576;           // 64
    float* gcv  = gsv + 64;                 // 64
    float* cmax = gcv + 64;                 // 8192 each
    float* csum = cmax + 8192;
    float* rmax = csum + 8192;
    float* rsum = rmax + 8192;
    float* s2mx = rsum + 8192;
    float* s2sm = s2mx + 8192;
    _Float16* xs1h = (_Float16*)(s2sm + 8192);  // 524,288 halves
    _Float16* xc1h = xs1h + 524288;             // 1,048,576
    _Float16* fsT  = xc1h + 1048576;            // 262,144
    _Float16* fcT  = fsT + 262144;              // 262,144

    dim3 T(256);
    // global context branch
    bcr_conv<<<1024, T, 0, stream>>>(xs, w_g1, nullptr, g_g1, nullptr, nullptr, ys, 2, 64, 0, 32, 64, 64);
    gap_kernel<<<64, T, 0, stream>>>(ys, gsv, out + OFF_GS);
    bcr_conv<<<1024, T, 0, stream>>>(xc, w_g2, nullptr, g_g2, nullptr, nullptr, yc, 2, 128, 0, 32, 64, 64);
    gap_kernel<<<64, T, 0, stream>>>(yc, gcv, out + OFF_GC);
    // local fusion: conv(concat(g,x)) + bias + BN + ReLU + residual x
    bcr_conv<<<2048, T, 0, stream>>>(xs, w_ls, b_ls, g_ls, gsv, xs, xs1, 2, 64, 32, 64, 64, 64);
    bcr_conv<<<4096, T, 0, stream>>>(xc, w_lc, b_lc, g_lc, gcv, xc, xc1, 2, 128, 32, 128, 64, 64);
    // attention features fs / fc (reuse ys/yc)
    bcr_conv<<<1024, T, 0, stream>>>(xs1, w_c1, nullptr, g_c1, nullptr, nullptr, ys, 2, 64, 0, 32, 64, 64);
    bcr_conv<<<1024, T, 0, stream>>>(xc1, w_c2, nullptr, g_c2, nullptr, nullptr, yc, 2, 128, 0, 32, 64, 64);
    // f16 operand prep
    transpose_f16<<<1024, T, 0, stream>>>(ys, fsT);
    transpose_f16<<<1024, T, 0, stream>>>(yc, fcT);
    cvt_f16<<<2048, T, 0, stream>>>(xs1, xs1h, 524288);
    cvt_f16<<<4096, T, 0, stream>>>(xc1, xc1h, 1048576);
    // att = fs^T fc  (WMMA, K=32)
    att_gemm<<<2048, T, 0, stream>>>(fsT, fcT, att);
    // softmax stats
    col_stats<<<32, T, 0, stream>>>(att, cmax, csum);
    row_stats<<<8192, T, 0, stream>>>(att, rmax, rsum);
    sym_stats<<<8192, T, 0, stream>>>(att, s2mx, s2sm);
    // fused softmax GEMMs (WMMA)
    out_gemm<4, false><<<64, T, 0, stream>>>(xs1h, att, cmax, csum, outs);
    out_gemm<8, true><<<64, T, 0, stream>>>(xc1h, att, rmax, rsum, outc);
    // output convs with residuals -> d_out
    bcr_conv<<<2048, T, 0, stream>>>(outs, w_o1, nullptr, g_o1, nullptr, xs1, out + OFF_OUTS, 2, 64, 0, 64, 64, 64);
    bcr_conv<<<4096, T, 0, stream>>>(outc, w_o2, nullptr, g_o2, nullptr, xc1, out + OFF_OUTC, 2, 128, 0, 128, 64, 64);
    // att -> att3 in place (symmetric pair tiles), then 1-ch 3x3 conv -> att_map
    att3_inplace<<<dim3(256, 256, 2), dim3(16, 16), 0, stream>>>(att, s2mx, s2sm);
    bcr_conv<<<131072, T, 0, stream>>>(att, w_oa, nullptr, g_oa, nullptr, nullptr, out, 2, 1, 0, 1, 4096, 4096);
}
